// STSelfAttention_61598420959649
// MI455X (gfx1250) — matrix operand order repaired and checked by reading, not compiled, not run on hardware
//
#include <hip/hip_runtime.h>
#include <hip/hip_bf16.h>

// ---------------------------------------------------------------------------
// Problem constants
// ---------------------------------------------------------------------------
#define BB     4
#define TT     48
#define NN     300
#define NPAD   304          // 19 * 16
#define NPADV  320          // key padding for PV K-chunks of 32
#define DD     320
#define HDIM   32
#define BT     (BB*TT)              // 192
#define MROWS  (BT*NPAD)            // 58368  (divisible by 256)
#define QKVC   704
#define SCALE_F 0.17677669529663687f   // 32^-0.5

// qkv column offsets
#define C_TQ 0
#define C_TK 64
#define C_TV 96
#define C_GQ 128
#define C_GK 256
#define C_GV 320
#define C_SQ 384
#define C_SK 448
#define C_SV 512
#define C_AQ 576
#define C_AK 640
#define C_AV 672
// concat column offsets
#define O_T   0
#define O_GEO 64
#define O_SEM 192
#define O_ADP 256
// Vt row offsets
#define VT_GEO 0
#define VT_SEM 64
#define VT_ADP 128
#define VT_ROWS 160

typedef _Float16 half_t;
typedef __attribute__((ext_vector_type(16))) _Float16 v16h;
typedef __attribute__((ext_vector_type(8)))  _Float16 h8;
typedef __attribute__((ext_vector_type(8)))  float    v8f;

// ---------------------------------------------------------------------------
// Async global->LDS copy (CDNA5: GLOBAL_LOAD_ASYNC_TO_LDS_B128, ASYNCcnt).
// LDS byte address = low 32 bits of the generic pointer on AMDGCN.
// ---------------------------------------------------------------------------
static __device__ __forceinline__ void async_copy16(void* lds_dst, const void* gsrc) {
  unsigned lds_off = (unsigned)(unsigned long long)(uintptr_t)lds_dst;
  unsigned long long ga = (unsigned long long)(uintptr_t)gsrc;
  asm volatile("global_load_async_to_lds_b128 %0, %1, off"
               :
               : "v"(lds_off), "v"(ga)
               : "memory");
}

static __device__ __forceinline__ void wait_async_lds() {
  asm volatile("s_wait_asynccnt 0x0" ::: "memory");
}

// ---------------------------------------------------------------------------
// WMMA helpers (CDNA5 gfx1250: wave32, D = A(16x32 f16) * B(32x16 f16) + C(f32))
// ---------------------------------------------------------------------------
static __device__ __forceinline__ v8f zero8() {
  v8f c;
#pragma unroll
  for (int i = 0; i < 8; ++i) c[i] = 0.0f;
  return c;
}

static __device__ __forceinline__ v8f wmma16(v16h a, v16h b, v8f c) {
  // v_wmma_f32_16x16x32_f16
  return __builtin_amdgcn_wmma_f32_16x16x32_f16(false, a, false, b, (short)0, c,
                                                false, false);
}

// A operand (16x32 f16, row-major source, K contiguous).
// Layout: lanes 0-15: M=lane, halves 0..15 = K0..15
//         lanes 16-31: M=lane-16, halves 0..15 = K16..31
static __device__ __forceinline__ v16h load_a16(const half_t* __restrict__ base,
                                                int stride, int lane) {
  int m = lane & 15;
  int sel = lane >> 4;
  const half_t* p = base + (size_t)m * stride + sel * 16;
  h8 lo = *(const h8*)(p);
  h8 hi = *(const h8*)(p + 8);
  v16h r;
#pragma unroll
  for (int i = 0; i < 8; ++i) { r[i] = lo[i]; r[8 + i] = hi[i]; }
  return r;
}

// B operand (32x16 f16) built from a row-major matrix W[n][k] (so B = W^T).
// Same K order as load_a16: lane holds column n = lane&15; lanes 0-15 hold K=0..15,
// lanes 16-31 hold K=16..31, contiguous halves.
static __device__ __forceinline__ v16h load_b16(const half_t* __restrict__ base,
                                                int stride, int lane) {
  int n = lane & 15;
  int kb = (lane >> 4) * 16;
  const half_t* p = base + (size_t)n * stride + kb;
  h8 lo = *(const h8*)(p);
  h8 hi = *(const h8*)(p + 8);
  v16h r;
#pragma unroll
  for (int i = 0; i < 8; ++i) { r[i] = lo[i]; r[8 + i] = hi[i]; }
  return r;
}

// C/D tile layout: vgpr i, lanes 0-15 -> (M=i, N=lane); lanes 16-31 -> (M=8+i)
static __device__ __forceinline__ void store_c_f16(half_t* __restrict__ dst,
                                                   int stride, v8f c, int lane) {
  int n = lane & 15;
  int rb = (lane >> 4) * 8;
#pragma unroll
  for (int i = 0; i < 8; ++i)
    dst[(size_t)(rb + i) * stride + n] = (half_t)c[i];
}

// ---------------------------------------------------------------------------
// Prep kernels
// ---------------------------------------------------------------------------
__global__ void prep_wh_kernel(const float* __restrict__ wtq, const float* __restrict__ wtk,
                               const float* __restrict__ wtv, const float* __restrict__ wgq,
                               const float* __restrict__ wgk, const float* __restrict__ wgv,
                               const float* __restrict__ wsq, const float* __restrict__ wsk,
                               const float* __restrict__ wsv, const float* __restrict__ waq,
                               const float* __restrict__ wak, const float* __restrict__ wav,
                               half_t* __restrict__ wh) {
  int idx = blockIdx.x * blockDim.x + threadIdx.x;
  if (idx >= QKVC * DD) return;
  int e = idx / DD, d = idx % DD;
  const float* src; int r;
  if      (e < 64)  { src = wtq; r = e; }
  else if (e < 96)  { src = wtk; r = e - 64; }
  else if (e < 128) { src = wtv; r = e - 96; }
  else if (e < 256) { src = wgq; r = e - 128; }
  else if (e < 320) { src = wgk; r = e - 256; }
  else if (e < 384) { src = wgv; r = e - 320; }
  else if (e < 448) { src = wsq; r = e - 384; }
  else if (e < 512) { src = wsk; r = e - 448; }
  else if (e < 576) { src = wsv; r = e - 512; }
  else if (e < 640) { src = waq; r = e - 576; }
  else if (e < 672) { src = wak; r = e - 640; }
  else              { src = wav; r = e - 672; }
  wh[(size_t)e * DD + d] = (half_t)src[(size_t)r * DD + d];
}

__global__ void prep_proj_kernel(const float* __restrict__ projw,
                                 half_t* __restrict__ wph) {
  int idx = blockIdx.x * blockDim.x + threadIdx.x;
  if (idx >= DD * DD) return;
  int e = idx / DD, d = idx % DD;
  wph[(size_t)e * DD + d] = (half_t)projw[(size_t)d * DD + e];  // transpose
}

__global__ void prep_x_kernel(const float* __restrict__ x, half_t* __restrict__ xh) {
  size_t idx = (size_t)blockIdx.x * blockDim.x + threadIdx.x;
  if (idx >= (size_t)MROWS * DD) return;
  size_t r = idx / DD; int d = (int)(idx % DD);
  int bt = (int)(r / NPAD), np = (int)(r % NPAD);
  float v = 0.0f;
  if (np < NN) v = x[(((size_t)bt * NN) + np) * DD + d];
  xh[idx] = (half_t)v;
}

// Vt[bt][row=hd-dim][col=key], zero padded to NPADV keys.
__global__ void prep_vt_kernel(const half_t* __restrict__ qkvh, half_t* __restrict__ vt) {
  size_t idx = (size_t)blockIdx.x * blockDim.x + threadIdx.x;
  if (idx >= (size_t)BT * VT_ROWS * NPADV) return;
  int bt = (int)(idx / (VT_ROWS * NPADV));
  int rem = (int)(idx % (VT_ROWS * NPADV));
  int vr = rem / NPADV, kk = rem % NPADV;
  int col = (vr < 64) ? (C_GV + vr) : (vr < 128) ? (C_SV + vr - 64) : (C_AV + vr - 128);
  half_t v = (half_t)0.0f;
  if (kk < NPAD) v = qkvh[((size_t)bt * NPAD + kk) * QKVC + col];
  vt[idx] = v;
}

// ---------------------------------------------------------------------------
// QKV projection GEMM: qkvh[MROWS x 704] = Xh[MROWS x 320] * Wh^T  (f16 out)
// block = 256 (8 waves); each wave computes a 32x64 C tile, software-pipelined
// so next K-chunk operand loads overlap the current chunk's 8 WMMAs.
// ---------------------------------------------------------------------------
__global__ __launch_bounds__(256) void gemm_qkv_kernel(const half_t* __restrict__ xh,
                                                       const half_t* __restrict__ wh,
                                                       half_t* __restrict__ qkvh) {
  int lane = threadIdx.x & 31;
  int wave = threadIdx.x >> 5;
  int row0 = (blockIdx.x * 8 + wave) * 32;
  int col0 = blockIdx.y * 64;
  const half_t* arow0 = xh + (size_t)row0 * DD;
  const half_t* arow1 = xh + (size_t)(row0 + 16) * DD;
  v8f acc[8];
#pragma unroll
  for (int i = 0; i < 8; ++i) acc[i] = zero8();

  v16h a0 = load_a16(arow0, DD, lane);
  v16h a1 = load_a16(arow1, DD, lane);
  v16h b[4];
#pragma unroll
  for (int nt = 0; nt < 4; ++nt)
    b[nt] = load_b16(wh + (size_t)(col0 + nt * 16) * DD, DD, lane);

#pragma unroll
  for (int kc = 0; kc < 10; ++kc) {
    v16h a0n, a1n, bn[4];
    if (kc < 9) {
      a0n = load_a16(arow0 + (kc + 1) * 32, DD, lane);
      a1n = load_a16(arow1 + (kc + 1) * 32, DD, lane);
#pragma unroll
      for (int nt = 0; nt < 4; ++nt)
        bn[nt] = load_b16(wh + (size_t)(col0 + nt * 16) * DD + (kc + 1) * 32, DD, lane);
    }
    if (kc < 8) {  // keep the A stream ahead in GL2 (global_prefetch_b8)
      __builtin_prefetch(arow0 + (kc + 2) * 32, 0, 1);
    }
#pragma unroll
    for (int nt = 0; nt < 4; ++nt) {
      acc[nt]     = wmma16(a0, b[nt], acc[nt]);
      acc[4 + nt] = wmma16(a1, b[nt], acc[4 + nt]);
    }
    if (kc < 9) {
      a0 = a0n; a1 = a1n;
#pragma unroll
      for (int nt = 0; nt < 4; ++nt) b[nt] = bn[nt];
    }
  }
#pragma unroll
  for (int rt = 0; rt < 2; ++rt)
#pragma unroll
    for (int nt = 0; nt < 4; ++nt)
      store_c_f16(qkvh + (size_t)(row0 + rt * 16) * QKVC + col0 + nt * 16, QKVC,
                  acc[rt * 4 + nt], lane);
}

// ---------------------------------------------------------------------------
// Output projection GEMM: out[.,e] = sum_d cat[.,d]*proj_w[d][e] + b[e] (f32)
// Wph is pre-transposed (Wph[e][d] = proj_w[d][e]). 32x64 tile per wave.
// ---------------------------------------------------------------------------
__global__ __launch_bounds__(256) void gemm_out_kernel(const half_t* __restrict__ cat,
                                                       const half_t* __restrict__ wph,
                                                       const float* __restrict__ bias,
                                                       float* __restrict__ out) {
  int lane = threadIdx.x & 31;
  int wave = threadIdx.x >> 5;
  int row0 = (blockIdx.x * 8 + wave) * 32;
  int col0 = blockIdx.y * 64;
  const half_t* arow0 = cat + (size_t)row0 * DD;
  const half_t* arow1 = cat + (size_t)(row0 + 16) * DD;
  v8f acc[8];
#pragma unroll
  for (int i = 0; i < 8; ++i) acc[i] = zero8();

  v16h a0 = load_a16(arow0, DD, lane);
  v16h a1 = load_a16(arow1, DD, lane);
  v16h b[4];
#pragma unroll
  for (int nt = 0; nt < 4; ++nt)
    b[nt] = load_b16(wph + (size_t)(col0 + nt * 16) * DD, DD, lane);

#pragma unroll
  for (int kc = 0; kc < 10; ++kc) {
    v16h a0n, a1n, bn[4];
    if (kc < 9) {
      a0n = load_a16(arow0 + (kc + 1) * 32, DD, lane);
      a1n = load_a16(arow1 + (kc + 1) * 32, DD, lane);
#pragma unroll
      for (int nt = 0; nt < 4; ++nt)
        bn[nt] = load_b16(wph + (size_t)(col0 + nt * 16) * DD + (kc + 1) * 32, DD, lane);
    }
#pragma unroll
    for (int nt = 0; nt < 4; ++nt) {
      acc[nt]     = wmma16(a0, b[nt], acc[nt]);
      acc[4 + nt] = wmma16(a1, b[nt], acc[4 + nt]);
    }
    if (kc < 9) {
      a0 = a0n; a1 = a1n;
#pragma unroll
      for (int nt = 0; nt < 4; ++nt) b[nt] = bn[nt];
    }
  }

  int n = lane & 15;
  int rb = (lane >> 4) * 8;
#pragma unroll
  for (int rt = 0; rt < 2; ++rt) {
#pragma unroll
    for (int nt = 0; nt < 4; ++nt) {
      int col = col0 + nt * 16 + n;
      float bcol = bias[col];
#pragma unroll
      for (int i = 0; i < 8; ++i) {
        int r = row0 + rt * 16 + rb + i;
        int bt = r / NPAD;
        int np = r - bt * NPAD;
        if (np < NN)
          out[(((size_t)bt * NN) + np) * DD + col] = acc[rt * 4 + nt][i] + bcol;
      }
    }
  }
}

// ---------------------------------------------------------------------------
// Spatial GQA attention (geo / sem / adaptive), WMMA everywhere.
// grid = (19 row tiles, H, BT), block = 32 (one wave).
// mask semantics: exclude = invert ? (mask==0) : (mask!=0)
// ---------------------------------------------------------------------------
__global__ __launch_bounds__(32) void spatial_attn_kernel(
    const half_t* __restrict__ qkvh, const half_t* __restrict__ vt,
    half_t* __restrict__ cat, const unsigned char* __restrict__ mask,
    int mask_bstride, int invert, int H, int G,
    int qoff, int koff, int vtoff, int outoff) {
  __shared__ float  sc[16][NPADV];     // raw scaled scores
  __shared__ half_t p16[16][NPADV];    // softmax probabilities (f16)

  int tile = blockIdx.x;
  int h    = blockIdx.y;
  int bt   = blockIdx.z;
  int lane = threadIdx.x;
  int g    = h / (H / G);
  int b    = bt / TT;

  // ---- scores: S = (Q K^T) * scale, 19 WMMAs, K-tile loads double-buffered --
  const half_t* qbase = qkvh + ((size_t)bt * NPAD + tile * 16) * QKVC + qoff + h * HDIM;
  const half_t* kcol  = qkvh + (size_t)bt * NPAD * QKVC + koff + g * HDIM;
  v16h aq = load_a16(qbase, QKVC, lane);
  int n  = lane & 15;
  int rb = (lane >> 4) * 8;
  v16h bk = load_b16(kcol, QKVC, lane);
#pragma unroll
  for (int kt = 0; kt < 19; ++kt) {
    v16h bkn;
    if (kt < 18) bkn = load_b16(kcol + (size_t)(kt + 1) * 16 * QKVC, QKVC, lane);
    v8f c = wmma16(aq, bk, zero8());
#pragma unroll
    for (int i = 0; i < 8; ++i) sc[rb + i][kt * 16 + n] = c[i] * SCALE_F;
    if (kt < 18) bk = bkn;
  }
  __syncthreads();

  // ---- masked softmax over each of the 16 rows (wave-cooperative) -----------
  for (int m = 0; m < 16; ++m) {
    int ig = tile * 16 + m;
    bool rowvalid = (ig < NN);
    float vals[10];
    float mx = -3.0e38f;
#pragma unroll
    for (int jj = 0; jj < 10; ++jj) {
      int j = lane + jj * 32;
      float v = -3.0e38f;
      if (rowvalid && j < NN) {
        unsigned char mm =
            mask[(size_t)(mask_bstride ? b * mask_bstride : 0) + (size_t)ig * NN + j];
        bool excl = invert ? (mm == 0) : (mm != 0);
        if (!excl) v = sc[m][j];
      }
      vals[jj] = v;
      mx = fmaxf(mx, v);
    }
#pragma unroll
    for (int off = 16; off > 0; off >>= 1) mx = fmaxf(mx, __shfl_xor(mx, off));
    float sum = 0.0f;
#pragma unroll
    for (int jj = 0; jj < 10; ++jj) {
      float e = (vals[jj] > -1.0e38f) ? __expf(vals[jj] - mx) : 0.0f;
      vals[jj] = e;
      sum += e;
    }
#pragma unroll
    for (int off = 16; off > 0; off >>= 1) sum += __shfl_xor(sum, off);
    float inv = 1.0f / (sum + 1.0e-20f);
#pragma unroll
    for (int jj = 0; jj < 10; ++jj) {
      int j = lane + jj * 32;
      p16[m][j] = (half_t)(vals[jj] * inv);
    }
  }
  __syncthreads();

  // ---- O = P V, 2 col-tiles x 10 K-chunks of WMMA (double-buffered) ---------
  const half_t* vtb = vt + (size_t)bt * VT_ROWS * NPADV;
#pragma unroll
  for (int ct = 0; ct < 2; ++ct) {
    const half_t* vrow = vtb + (size_t)(vtoff + g * HDIM + ct * 16) * NPADV;
    v8f acc = zero8();
    v16h ap = load_a16(&p16[0][0], NPADV, lane);
    v16h bv = load_b16(vrow, NPADV, lane);
#pragma unroll
    for (int kc = 0; kc < 10; ++kc) {
      v16h apn, bvn;
      if (kc < 9) {
        apn = load_a16(&p16[0][0] + (kc + 1) * 32, NPADV, lane);
        bvn = load_b16(vrow + (kc + 1) * 32, NPADV, lane);
      }
      acc = wmma16(ap, bv, acc);
      if (kc < 9) { ap = apn; bv = bvn; }
    }
#pragma unroll
    for (int i = 0; i < 8; ++i) {
      int ig = tile * 16 + rb + i;
      if (ig < NN)
        cat[((size_t)bt * NPAD + ig) * DD + outoff + h * HDIM + ct * 16 + n] =
            (half_t)acc[i];
    }
  }
}

// ---------------------------------------------------------------------------
// Temporal MQA branch (T=48, flash-style, VALU). grid = B*N, block = 64.
// K/V tiles staged to LDS with GLOBAL_LOAD_ASYNC_TO_LDS_B128 (inline asm).
// ---------------------------------------------------------------------------
__global__ __launch_bounds__(64) void temporal_attn_kernel(
    const half_t* __restrict__ qkvh, half_t* __restrict__ cat) {
  __shared__ half_t ks[TT][HDIM];
  __shared__ half_t vs[TT][HDIM];
  int bn = blockIdx.x;
  int b = bn / NN, nnode = bn % NN;
  int tid = threadIdx.x;
  // 48 rows x (4 K-chunks + 4 V-chunks) of 16B each, async global->LDS
  for (int idx = tid; idx < TT * 8; idx += 64) {
    int s = idx >> 3;
    int c = idx & 7;
    size_t row = ((size_t)b * TT + s) * NPAD + nnode;
    const half_t* src =
        qkvh + row * QKVC + (c < 4 ? (C_TK + c * 8) : (C_TV + (c - 4) * 8));
    half_t* dst = (c < 4) ? &ks[s][c * 8] : &vs[s][(c - 4) * 8];
    async_copy16(dst, src);
  }
  wait_async_lds();
  __syncthreads();
  if (tid >= TT) return;
  int t = tid;
  size_t qrow = ((size_t)b * TT + t) * NPAD + nnode;
  for (int h = 0; h < 2; ++h) {
    float q[HDIM];
#pragma unroll
    for (int d = 0; d < HDIM; ++d)
      q[d] = (float)qkvh[qrow * QKVC + C_TQ + h * HDIM + d];
    float mrun = -3.0e38f, lrun = 0.0f, o[HDIM];
#pragma unroll
    for (int d = 0; d < HDIM; ++d) o[d] = 0.0f;
    for (int s = 0; s < TT; ++s) {
      float sdot = 0.0f;
#pragma unroll
      for (int d = 0; d < HDIM; ++d) sdot += q[d] * (float)ks[s][d];
      sdot *= SCALE_F;
      float mn = fmaxf(mrun, sdot);
      float corr = __expf(mrun - mn);
      float p = __expf(sdot - mn);
      lrun = lrun * corr + p;
#pragma unroll
      for (int d = 0; d < HDIM; ++d) o[d] = o[d] * corr + p * (float)vs[s][d];
      mrun = mn;
    }
    float inv = 1.0f / lrun;
#pragma unroll
    for (int d = 0; d < HDIM; ++d)
      cat[qrow * DD + O_T + h * HDIM + d] = (half_t)(o[d] * inv);
  }
}

// ---------------------------------------------------------------------------
// Adaptive-graph chain (small VALU kernels)
// ---------------------------------------------------------------------------
__global__ void mean_x_kernel(const float* __restrict__ x, float* __restrict__ xm) {
  int bn = blockIdx.x;             // b*N + n
  int b = bn / NN, nnode = bn % NN;
  int d = threadIdx.x;             // 320
  float s = 0.0f;
  for (int t = 0; t < TT; ++t)
    s += x[(((size_t)b * TT + t) * NN + nnode) * DD + d];
  xm[(size_t)bn * DD + d] = s * (1.0f / TT);
}

__global__ void mlp1_kernel(const float* __restrict__ xm, const float* __restrict__ g1w,
                            const float* __restrict__ g1b, const float* __restrict__ lng,
                            const float* __restrict__ lnb, float* __restrict__ hbuf) {
  __shared__ float hs[160];
  int bn = blockIdx.x;
  int e = threadIdx.x;             // 160
  const float* xr = xm + (size_t)bn * DD;
  float v = g1b[e];
  for (int d = 0; d < DD; ++d) v += xr[d] * g1w[(size_t)d * 160 + e];
  hs[e] = v;
  __syncthreads();
  float mu = 0.0f, m2 = 0.0f;
  for (int j = 0; j < 160; ++j) { mu += hs[j]; m2 += hs[j] * hs[j]; }
  mu *= (1.0f / 160.0f);
  float var = m2 * (1.0f / 160.0f) - mu * mu;
  float hn = (v - mu) * rsqrtf(var + 1e-5f) * lng[e] + lnb[e];
  // tanh-approx gelu (jax.nn.gelu default)
  float c0 = 0.7978845608028654f;
  float inner = c0 * (hn + 0.044715f * hn * hn * hn);
  float gelu = 0.5f * hn * (1.0f + tanhf(inner));
  hbuf[(size_t)bn * 160 + e] = gelu;
}

__global__ void mlp2_kernel(const float* __restrict__ hbuf, const float* __restrict__ g2w,
                            const float* __restrict__ g2b, float* __restrict__ gbuf) {
  __shared__ float gs[80];
  int bn = blockIdx.x;
  int e = threadIdx.x;             // 80
  const float* hr = hbuf + (size_t)bn * 160;
  float v = g2b[e];
  for (int d = 0; d < 160; ++d) v += hr[d] * g2w[(size_t)d * 80 + e];
  gs[e] = v;
  __syncthreads();
  float ss = 0.0f;
  for (int j = 0; j < 80; ++j) ss += gs[j] * gs[j];
  float rn = 1.0f / (sqrtf(ss) + 1e-8f);
  gbuf[(size_t)bn * 80 + e] = v * rn;
}

__global__ __launch_bounds__(64) void topk_kernel(const float* __restrict__ gbuf,
                                                  unsigned char* __restrict__ keep) {
  __shared__ float sr[NPAD];
  int bn = blockIdx.x;
  int b = bn / NN, i = bn % NN;
  int tid = threadIdx.x;
  const float* gi = gbuf + ((size_t)b * NN + i) * 80;
  for (int j = tid; j < NPAD; j += 64) {
    float v = -3.0e38f;
    if (j < NN) {
      const float* gj = gbuf + ((size_t)b * NN + j) * 80;
      float s = 0.0f;
#pragma unroll 8
      for (int d = 0; d < 80; ++d) s += gi[d] * gj[d];
      v = s;
      keep[(size_t)b * NN * NN + (size_t)i * NN + j] = 0;
    }
    sr[j] = v;
  }
  __syncthreads();
  if (tid == 0) {
    for (int k = 0; k < 10; ++k) {
      int best = 0;
      float bv = sr[0];
      for (int j = 1; j < NN; ++j)
        if (sr[j] > bv) { bv = sr[j]; best = j; }   // strict > -> lowest index on ties
      keep[(size_t)b * NN * NN + (size_t)i * NN + best] = 1;
      sr[best] = -3.0e38f;
    }
  }
}

// ---------------------------------------------------------------------------
// Host launcher
// ---------------------------------------------------------------------------
extern "C" void kernel_launch(void* const* d_in, const int* in_sizes, int n_in,
                              void* d_out, int out_size, void* d_ws, size_t ws_size,
                              hipStream_t stream) {
  (void)in_sizes; (void)n_in; (void)out_size; (void)ws_size;
  const float* x     = (const float*)d_in[0];
  const unsigned char* geo_mask = (const unsigned char*)d_in[1];
  const unsigned char* sem_mask = (const unsigned char*)d_in[2];
  const float* wtq = (const float*)d_in[3];
  const float* wtk = (const float*)d_in[4];
  const float* wtv = (const float*)d_in[5];
  const float* wgq = (const float*)d_in[6];
  const float* wgk = (const float*)d_in[7];
  const float* wgv = (const float*)d_in[8];
  const float* wsq = (const float*)d_in[9];
  const float* wsk = (const float*)d_in[10];
  const float* wsv = (const float*)d_in[11];
  const float* waq = (const float*)d_in[12];
  const float* wak = (const float*)d_in[13];
  const float* wav = (const float*)d_in[14];
  const float* g1w = (const float*)d_in[15];
  const float* g1b = (const float*)d_in[16];
  const float* lng = (const float*)d_in[17];
  const float* lnb = (const float*)d_in[18];
  const float* g2w = (const float*)d_in[19];
  const float* g2b = (const float*)d_in[20];
  const float* projw = (const float*)d_in[21];
  const float* projb = (const float*)d_in[22];
  float* out = (float*)d_out;

  // -------- workspace carving (256B aligned) --------
  char* ws = (char*)d_ws;
  size_t off = 0;
  auto carve = [&](size_t bytes) {
    size_t r = off;
    off = (off + bytes + 255) & ~(size_t)255;
    return r;
  };
  half_t* xh   = (half_t*)(ws + carve((size_t)MROWS * DD * 2));
  half_t* qkvh = (half_t*)(ws + carve((size_t)MROWS * QKVC * 2));
  half_t* vt   = (half_t*)(ws + carve((size_t)BT * VT_ROWS * NPADV * 2));
  half_t* cat  = (half_t*)(ws + carve((size_t)MROWS * DD * 2));
  half_t* wh   = (half_t*)(ws + carve((size_t)QKVC * DD * 2));
  half_t* wph  = (half_t*)(ws + carve((size_t)DD * DD * 2));
  float*  xm   = (float*)(ws + carve((size_t)BB * NN * DD * 4));
  float*  hbuf = (float*)(ws + carve((size_t)BB * NN * 160 * 4));
  float*  gbuf = (float*)(ws + carve((size_t)BB * NN * 80 * 4));
  unsigned char* keep = (unsigned char*)(ws + carve((size_t)BB * NN * NN));

  // -------- prep --------
  {
    int tot = QKVC * DD;
    prep_wh_kernel<<<(tot + 255) / 256, 256, 0, stream>>>(
        wtq, wtk, wtv, wgq, wgk, wgv, wsq, wsk, wsv, waq, wak, wav, wh);
  }
  {
    int tot = DD * DD;
    prep_proj_kernel<<<(tot + 255) / 256, 256, 0, stream>>>(projw, wph);
  }
  {
    size_t tot = (size_t)MROWS * DD;
    prep_x_kernel<<<(unsigned)((tot + 255) / 256), 256, 0, stream>>>(x, xh);
  }

  // -------- QKV projection (WMMA, 32x64 tile/wave, pipelined) --------
  gemm_qkv_kernel<<<dim3(MROWS / 256, QKVC / 64), 256, 0, stream>>>(xh, wh, qkvh);

  // -------- V transpose for PV stage --------
  {
    size_t tot = (size_t)BT * VT_ROWS * NPADV;
    prep_vt_kernel<<<(unsigned)((tot + 255) / 256), 256, 0, stream>>>(qkvh, vt);
  }

  // -------- adaptive-graph chain --------
  mean_x_kernel<<<BB * NN, DD, 0, stream>>>(x, xm);
  mlp1_kernel<<<BB * NN, 160, 0, stream>>>(xm, g1w, g1b, lng, lnb, hbuf);
  mlp2_kernel<<<BB * NN, 80, 0, stream>>>(hbuf, g2w, g2b, gbuf);
  topk_kernel<<<BB * NN, 64, 0, stream>>>(gbuf, keep);

  // -------- temporal branch --------
  temporal_attn_kernel<<<BB * NN, 64, 0, stream>>>(qkvh, cat);

  // -------- spatial branches (WMMA attention) --------
  // geo: H=4, G=2
  spatial_attn_kernel<<<dim3(19, 4, BT), 32, 0, stream>>>(
      qkvh, vt, cat, geo_mask, 0, 0, 4, 2, C_GQ, C_GK, VT_GEO, O_GEO);
  // sem: H=2, G=2
  spatial_attn_kernel<<<dim3(19, 2, BT), 32, 0, stream>>>(
      qkvh, vt, cat, sem_mask, 0, 0, 2, 2, C_SQ, C_SK, VT_SEM, O_SEM);
  // adp: H=2, G=1, keep-mask with inverted semantics
  spatial_attn_kernel<<<dim3(19, 2, BT), 32, 0, stream>>>(
      qkvh, vt, cat, keep, NN * NN, 1, 2, 1, C_AQ, C_AK, VT_ADP, O_ADP);

  // -------- output projection (WMMA) --------
  gemm_out_kernel<<<dim3(MROWS / 256, DD / 64), 256, 0, stream>>>(cat, wph, projb, out);
}
